// instance_contrastive_Loss_45930380264015
// MI455X (gfx1250) — matrix operand order repaired and checked by reading, not compiled, not run on hardware
//
#include <hip/hip_runtime.h>
#include <hip/hip_bf16.h>

// ---------------------------------------------------------------------------
// Supervised contrastive loss, N=8192 D=128 C=64, f16 WMMA path for gfx1250.
// - rows pre-scaled by sqrt(2) so WMMA output is already 2*cos_sim (temp fold)
// - ping-pong pipelined B-tile loads overlapped with WMMA
// - diagonal handled once per wave in a uniform branch, not per tile
// ---------------------------------------------------------------------------

typedef __attribute__((ext_vector_type(16))) _Float16 v16h;
typedef __attribute__((ext_vector_type(8)))  _Float16 v8h;
typedef __attribute__((ext_vector_type(8)))  float    v8f;

#define N_ROWS 8192
#define DIM    128
#define SQRT2  1.41421356237309515f   // folds 1/T=2 into the data: (√2x)·(√2y)=2x·y

// ---------------------------------------------------------------------------
// Zero the 64-entry label histogram.
__global__ void k_init_hist(int* __restrict__ hist) {
    hist[threadIdx.x] = 0;
}

// ---------------------------------------------------------------------------
// Row L2-normalize f32 -> f16 scaled by sqrt(2), and build label histogram.
// One block (128 threads) per row.
__global__ void __launch_bounds__(128) k_normalize(
    const float* __restrict__ x, const int* __restrict__ label,
    _Float16* __restrict__ xh, int* __restrict__ hist) {
    const int row = blockIdx.x;
    const int tid = threadIdx.x;
    const float v = x[(size_t)row * DIM + tid];
    __shared__ float red[128];
    red[tid] = v * v;
    __syncthreads();
    #pragma unroll
    for (int s = 64; s > 0; s >>= 1) {
        if (tid < s) red[tid] += red[tid + s];
        __syncthreads();
    }
    const float inv = SQRT2 / fmaxf(__builtin_sqrtf(red[0]), 1e-12f);
    xh[(size_t)row * DIM + tid] = (_Float16)(v * inv);
    if (tid == 0) atomicAdd(&hist[label[row]], 1);
}

// ---------------------------------------------------------------------------
// Core: sim = exp(x~ x~^T) with x~ = sqrt(2)*x_hat, fused masked row sums.
// Each wave owns a 16-row strip; block of 4 waves covers 64 rows; 128 blocks
// cover N=8192. Every row is reduced entirely inside one wave ->
// deterministic, no float atomics.
__global__ void __launch_bounds__(128) k_sim(
    const _Float16* __restrict__ xh, const int* __restrict__ label,
    float* __restrict__ negS, float* __restrict__ posS) {
    const int tid  = threadIdx.x;
    const int wave = tid >> 5;
    const int lane = tid & 31;
    const int lo   = lane & 15;   // N / M position within 16
    const int hi   = lane >> 4;   // half-wave selector
    const int i_base = blockIdx.x * 64 + wave * 16;

    // --- A fragments: 16 rows x 128 K, resident for the whole sweep.
    // A 16x32 layout (matches B): lanes 0-15 hold K {0..15} of the chunk,
    // lanes 16-31 hold K {16..31}.
    const _Float16* arow = xh + (size_t)(i_base + lo) * DIM;
    v16h a[4];
    #pragma unroll
    for (int kc = 0; kc < 4; ++kc) {
        const v8h a0 = *(const v8h*)(arow + kc * 32 + hi * 16);
        const v8h a1 = *(const v8h*)(arow + kc * 32 + hi * 16 + 8);
        a[kc] = __builtin_shufflevector(a0, a1,
                                        0, 1, 2, 3, 4, 5, 6, 7,
                                        8, 9, 10, 11, 12, 13, 14, 15);
    }

    // Labels of this wave's rows, per the C-tile layout (VGPR v -> row v+8*hi).
    int labr[8];
    #pragma unroll
    for (int v = 0; v < 8; ++v) labr[v] = label[i_base + v + 8 * hi];

    // totAcc: sum over ALL columns (diagonal included, removed at the end).
    // posAcc: sum over same-label columns (diagonal included, removed at end,
    //         valid since label[i]==label[i]).
    // diag  : this row's own sim value, captured in the one diagonal tile.
    float totAcc[8], posAcc[8], diag[8];
    #pragma unroll
    for (int v = 0; v < 8; ++v) { totAcc[v] = 0.0f; posAcc[v] = 0.0f; diag[v] = 0.0f; }

    // B-tile loader: 32x16 f16 tile = columns j..j+15 of x^T = rows j..j+15
    // of x. Lane (hi,lo): 16 contiguous halfs of row (j+lo), K offset hi*16.
    auto loadB = [&](v16h (&b)[4], int& labc, int j) {
        const _Float16* brow = xh + (size_t)(j + lo) * DIM + hi * 16;
        #pragma unroll
        for (int kc = 0; kc < 4; ++kc)
            b[kc] = *(const v16h*)(brow + kc * 32);
        labc = label[j + lo];
    };

    // Consume one 16x16 tile at column base j0: 4 chained WMMAs over K=128,
    // native-exp (arg bounded in [-2,2]) + masked accumulation. Diagonal is
    // captured only in the single tile where j0 == i_base (scalar-uniform
    // branch; EXEC untouched).
    auto computeTile = [&](const v16h (&b)[4], int labc, int j0) {
        v8f c = {};
        #pragma unroll
        for (int kc = 0; kc < 4; ++kc)
            c = __builtin_amdgcn_wmma_f32_16x16x32_f16(
                    false, a[kc], false, b[kc], (short)0, c, false, false);
        float sv[8];
        #pragma unroll
        for (int v = 0; v < 8; ++v) {
            sv[v] = __expf(c[v]);                     // temp already folded in
            totAcc[v] += sv[v];
            posAcc[v] += (labr[v] == labc) ? sv[v] : 0.0f;
        }
        if (j0 == i_base) {                            // uniform: diagonal tile
            #pragma unroll
            for (int v = 0; v < 8; ++v)
                diag[v] += ((v + 8 * hi) == lo) ? sv[v] : 0.0f;
        }
    };

    // --- Ping-pong pipelined sweep over all 512 column tiles (even count).
    v16h b0[4], b1[4];
    int  l0, l1;
    loadB(b0, l0, 0);
    for (int j0 = 0; j0 < N_ROWS; j0 += 32) {
        // Warm L2 two tiles ahead (wraps harmlessly near the end).
        __builtin_prefetch(
            xh + (size_t)(((j0 + 64) & (N_ROWS - 1)) + lo) * DIM + hi * 16,
            0, 0);
        loadB(b1, l1, (j0 + 16) & (N_ROWS - 1));   // next tile while computing
        computeTile(b0, l0, j0);
        loadB(b0, l0, (j0 + 32) & (N_ROWS - 1));   // tile after next
        computeTile(b1, l1, j0 + 16);
    }

    // --- Remove diagonal, reduce across the 16 lanes of each half-wave;
    // one lane writes each row.
    #pragma unroll
    for (int v = 0; v < 8; ++v) {
        float n = totAcc[v] - diag[v];
        float p = posAcc[v] - diag[v];
        #pragma unroll
        for (int m = 8; m >= 1; m >>= 1) {
            n += __shfl_xor(n, m, 32);
            p += __shfl_xor(p, m, 32);
        }
        if (lo == 0) {
            const int row = i_base + v + 8 * hi;
            negS[row] = n;
            posS[row] = p;
        }
    }
}

// ---------------------------------------------------------------------------
// Final scalar: mean over rows of log(neg) - log(pos/cnt).
__global__ void __launch_bounds__(256) k_loss(
    const float* __restrict__ negS, const float* __restrict__ posS,
    const int* __restrict__ label, const int* __restrict__ hist,
    float* __restrict__ out) {
    __shared__ float red[256];
    const int tid = threadIdx.x;
    float acc = 0.0f;
    for (int r = tid; r < N_ROWS; r += 256) {
        const float cnt = (float)(hist[label[r]] - 1);
        const float pos = posS[r] / cnt;
        acc += __builtin_logf(negS[r]) - __builtin_logf(pos);
    }
    red[tid] = acc;
    __syncthreads();
    #pragma unroll
    for (int s = 128; s > 0; s >>= 1) {
        if (tid < s) red[tid] += red[tid + s];
        __syncthreads();
    }
    if (tid == 0) out[0] = red[0] / (float)N_ROWS;
}

// ---------------------------------------------------------------------------
extern "C" void kernel_launch(void* const* d_in, const int* in_sizes, int n_in,
                              void* d_out, int out_size, void* d_ws, size_t ws_size,
                              hipStream_t stream) {
    const float* x     = (const float*)d_in[0];
    const int*   label = (const int*)d_in[1];
    float*       out   = (float*)d_out;

    // Workspace layout:
    //   [0, 2 MB)            : x~ = sqrt(2) * x_hat as f16 (8192 x 128)
    //   [2 MB, +32 KB)       : negS  (per-row off-diagonal sum)
    //   [.., +32 KB)         : posS  (per-row same-label off-diagonal sum)
    //   [.., +256 B)         : hist  (64 class counts)
    char* ws = (char*)d_ws;
    _Float16* xh   = (_Float16*)ws;
    float*    negS = (float*)(ws + (size_t)N_ROWS * DIM * sizeof(_Float16));
    float*    posS = negS + N_ROWS;
    int*      hist = (int*)(posS + N_ROWS);

    k_init_hist<<<1, 64, 0, stream>>>(hist);
    k_normalize<<<N_ROWS, 128, 0, stream>>>(x, label, xh, hist);
    k_sim<<<N_ROWS / 64, 128, 0, stream>>>(xh, label, negS, posS);
    k_loss<<<1, 256, 0, stream>>>(negS, posS, label, hist, out);
}
